// GAT_26087631356717
// MI455X (gfx1250) — matrix operand order repaired and checked by reading, not compiled, not run on hardware
//
#include <hip/hip_runtime.h>

typedef __attribute__((ext_vector_type(16))) _Float16 v16h;
typedef __attribute__((ext_vector_type(8)))  float    v8f;

#define NNODES   100000
#define NEDGES   1600000
#define FDIM     128          // F == H*D == 128 for both layers
#define NHEAD    4
#define NEG_SLOPE 0.2f
#define MTILES   (NNODES / 16)   // 6250, exact

// ---------------------------------------------------------------------------
// Pack fp32 weight matrix [128][128] into WMMA B-tile layout as fp16.
// Flat index t = ((kt*8+nt)*32 + lane)*16 + j  ->  W[kt*32 + (j>>3)*16 + (lane>>4)*8 + (j&7)][nt*16 + (lane&15)]
// ---------------------------------------------------------------------------
__global__ void pack_w_kernel(const float* __restrict__ W, _Float16* __restrict__ Wp) {
    int t    = blockIdx.x * 256 + threadIdx.x;      // 16384 total, grid exact
    int j    = t & 15;
    int lane = (t >> 4) & 31;
    int nt   = (t >> 9) & 7;
    int kt   = t >> 12;
    int k    = kt * 32 + (j >> 3) * 16 + (lane >> 4) * 8 + (j & 7);
    int n    = nt * 16 + (lane & 15);
    Wp[t] = (_Float16)W[k * FDIM + n];
}

// ---------------------------------------------------------------------------
// GEMM: Hout[N,128] = X[N,128] @ W[128,128]  (bias added later in aggregation)
// One wave per 16-row strip; 8 v8f accumulators cover all 128 output columns.
// ---------------------------------------------------------------------------
__global__ __launch_bounds__(128)
void gemm_wmma_kernel(const float* __restrict__ X, const _Float16* __restrict__ Wp,
                      float* __restrict__ Hout) {
    __shared__ float ldsbuf[4 * 16 * FDIM];         // 32 KB, one 8KB slab per wave

    int wave = threadIdx.x >> 5;
    int lane = threadIdx.x & 31;
    int mt   = blockIdx.x * 4 + wave;
    bool active = (mt < MTILES);
    if (!active) mt = 0;                            // compute on valid memory, skip stores

    int  kgrp = lane >> 4;                          // 0 or 1
    const float* xrow = X + (size_t)(mt * 16 + (lane & 15)) * FDIM;

    v8f zero = {};
    v8f acc[8];
#pragma unroll
    for (int nt = 0; nt < 8; ++nt) acc[nt] = zero;

#pragma unroll
    for (int kt = 0; kt < 4; ++kt) {
        if (kt < 3) __builtin_prefetch(xrow + (kt + 1) * 32 + kgrp * 8, 0, 0);

        // A fragment: ISA 16-bit A 16x32 layout.
        // lane<16: halves 0..7 = K 0..7, halves 8..15 = K 16..23 (kbase=0)
        // lane>=16: kbase=8
        const float4* p0 = (const float4*)(xrow + kt * 32 + kgrp * 8);
        const float4* p1 = (const float4*)(xrow + kt * 32 + 16 + kgrp * 8);
        float4 a0 = p0[0], a1 = p0[1], a2 = p1[0], a3 = p1[1];
        v16h a;
        a[0]  = (_Float16)a0.x; a[1]  = (_Float16)a0.y;
        a[2]  = (_Float16)a0.z; a[3]  = (_Float16)a0.w;
        a[4]  = (_Float16)a1.x; a[5]  = (_Float16)a1.y;
        a[6]  = (_Float16)a1.z; a[7]  = (_Float16)a1.w;
        a[8]  = (_Float16)a2.x; a[9]  = (_Float16)a2.y;
        a[10] = (_Float16)a2.z; a[11] = (_Float16)a2.w;
        a[12] = (_Float16)a3.x; a[13] = (_Float16)a3.y;
        a[14] = (_Float16)a3.z; a[15] = (_Float16)a3.w;

        const v16h* btiles = (const v16h*)Wp + (size_t)(kt * 8) * 32 + lane;
#pragma unroll
        for (int nt = 0; nt < 8; ++nt) {
            v16h b = btiles[nt * 32];
            acc[nt] = __builtin_amdgcn_wmma_f32_16x16x32_f16(
                false, a, false, b, (short)0, acc[nt], false, false);
        }
    }

    // C layout: acc[nt] VGPR r, lane l -> row = r + (l>=16?8:0), col = nt*16 + (l&15).
    // Stage through LDS for fully coalesced 512B row stores.
    float* lw = ldsbuf + wave * (16 * FDIM);
    int colbase = lane & 15;
    int rowoff  = kgrp * 8;
#pragma unroll
    for (int nt = 0; nt < 8; ++nt)
#pragma unroll
        for (int r = 0; r < 8; ++r)
            lw[(rowoff + r) * FDIM + nt * 16 + colbase] = acc[nt][r];
    __builtin_amdgcn_wave_barrier();                // wave-local: DS ops in-order per wave

    if (active) {
#pragma unroll
        for (int r16 = 0; r16 < 16; ++r16) {
            float4 v = *(const float4*)(lw + r16 * FDIM + lane * 4);
            *(float4*)(Hout + (size_t)(mt * 16 + r16) * FDIM + lane * 4) = v;
        }
    }
}

// ---------------------------------------------------------------------------
// Per-node attention terms: el[n,h] = <h[n,h,:], al[h,:]>, er likewise.
// One wave per node; lane covers 4 contiguous cols; head = lane>>3.
// ---------------------------------------------------------------------------
__global__ __launch_bounds__(256)
void attn_terms_kernel(const float* __restrict__ Hl, const float* __restrict__ al,
                       const float* __restrict__ ar, float* __restrict__ el,
                       float* __restrict__ er) {
    int wave = threadIdx.x >> 5, lane = threadIdx.x & 31;
    int n = blockIdx.x * 8 + wave;
    if (n >= NNODES) return;
    float4 h4 = *(const float4*)(Hl + (size_t)n * FDIM + lane * 4);
    float4 l4 = *(const float4*)(al + lane * 4);
    float4 r4 = *(const float4*)(ar + lane * 4);
    float sl = h4.x * l4.x + h4.y * l4.y + h4.z * l4.z + h4.w * l4.w;
    float sr = h4.x * r4.x + h4.y * r4.y + h4.z * r4.z + h4.w * r4.w;
#pragma unroll
    for (int off = 1; off < 8; off <<= 1) {
        sl += __shfl_xor(sl, off, 32);
        sr += __shfl_xor(sr, off, 32);
    }
    if ((lane & 7) == 0) {
        int hh = lane >> 3;
        el[n * NHEAD + hh] = sl;
        er[n * NHEAD + hh] = sr;
    }
}

// ---------------------------------------------------------------------------
// Per-layer init: out = bias (broadcast), maxflip = 0 (below flip(-inf)), denom = 0
// ---------------------------------------------------------------------------
__global__ void init_layer_kernel(float* __restrict__ out, const float* __restrict__ bias,
                                  unsigned* __restrict__ maxf, float* __restrict__ denom) {
    int t = blockIdx.x * 256 + threadIdx.x;         // grid covers NNODES*FDIM exactly
    out[t] = bias[t & (FDIM - 1)];
    if (t < NNODES * NHEAD) { maxf[t] = 0u; denom[t] = 0.0f; }
}

// ---------------------------------------------------------------------------
// Pass 1: e = leakyrelu(el[src]+er[dst]); store e; segment-max via ordered-uint atomicMax
// ---------------------------------------------------------------------------
__global__ void edge_logit_max_kernel(const int* __restrict__ src, const int* __restrict__ dst,
                                      const float* __restrict__ el, const float* __restrict__ er,
                                      float* __restrict__ ebuf, unsigned* __restrict__ maxf) {
    int t = blockIdx.x * 256 + threadIdx.x;         // E*H exact
    int e = t >> 2, hh = t & 3;
    int s = src[e], d = dst[e];
    float v = el[s * NHEAD + hh] + er[d * NHEAD + hh];
    v = v > 0.0f ? v : v * NEG_SLOPE;
    ebuf[t] = v;
    unsigned u = __float_as_uint(v);
    u = (u & 0x80000000u) ? ~u : (u | 0x80000000u); // monotone float->uint map
    atomicMax(maxf + d * NHEAD + hh, u);
}

// ---------------------------------------------------------------------------
// Pass 2: a = exp(e - m[dst]); store a in-place; denom[dst] += a
// ---------------------------------------------------------------------------
__global__ void edge_exp_sum_kernel(const int* __restrict__ dst, float* __restrict__ ebuf,
                                    const unsigned* __restrict__ maxf, float* __restrict__ denom) {
    int t = blockIdx.x * 256 + threadIdx.x;
    int e = t >> 2, hh = t & 3;
    int d = dst[e];
    unsigned u = maxf[d * NHEAD + hh];
    float m = (u & 0x80000000u) ? __uint_as_float(u & 0x7fffffffu) : __uint_as_float(~u);
    float a = __expf(ebuf[t] - m);
    ebuf[t] = a;
    atomicAdd(denom + d * NHEAD + hh, a);
}

// ---------------------------------------------------------------------------
// Pass 3: out[dst] += h[src] * (a / max(denom[dst],1e-9)).  One wave per edge:
// 32 lanes x float4 = full 128-col row; head = lane>>3.
// ---------------------------------------------------------------------------
__global__ __launch_bounds__(256)
void edge_aggregate_kernel(const int* __restrict__ src, const int* __restrict__ dst,
                           const float* __restrict__ Hl, const float* __restrict__ ebuf,
                           const float* __restrict__ denom, float* __restrict__ out) {
    int wave = threadIdx.x >> 5, lane = threadIdx.x & 31;
    int e = blockIdx.x * 8 + wave;                  // E/8 exact
    int s = src[e], d = dst[e];
    int hh = lane >> 3;
    float a   = ebuf[e * NHEAD + hh];
    float den = denom[d * NHEAD + hh];
    float alpha = a / fmaxf(den, 1e-9f);
    float4 h4 = *(const float4*)(Hl + (size_t)s * FDIM + lane * 4);
    float* o = out + (size_t)d * FDIM + lane * 4;
    atomicAdd(o + 0, h4.x * alpha);
    atomicAdd(o + 1, h4.y * alpha);
    atomicAdd(o + 2, h4.z * alpha);
    atomicAdd(o + 3, h4.w * alpha);
}

// ---------------------------------------------------------------------------
// Final classifier: y[n,c] = <h[n,:], Wf[:,c]> + bf[c], C=2. One wave per node.
// ---------------------------------------------------------------------------
__global__ __launch_bounds__(256)
void classifier_kernel(const float* __restrict__ Hin, const float* __restrict__ Wf,
                       const float* __restrict__ bf, float* __restrict__ y) {
    int wave = threadIdx.x >> 5, lane = threadIdx.x & 31;
    int n = blockIdx.x * 8 + wave;
    if (n >= NNODES) return;
    float4 h4 = *(const float4*)(Hin + (size_t)n * FDIM + lane * 4);
    float hv[4] = {h4.x, h4.y, h4.z, h4.w};
    float s0 = 0.0f, s1 = 0.0f;
#pragma unroll
    for (int j = 0; j < 4; ++j) {
        s0 += hv[j] * Wf[(lane * 4 + j) * 2 + 0];
        s1 += hv[j] * Wf[(lane * 4 + j) * 2 + 1];
    }
#pragma unroll
    for (int off = 1; off < 32; off <<= 1) {
        s0 += __shfl_xor(s0, off, 32);
        s1 += __shfl_xor(s1, off, 32);
    }
    if (lane == 0) {
        y[n * 2 + 0] = s0 + bf[0];
        y[n * 2 + 1] = s1 + bf[1];
    }
}

// ---------------------------------------------------------------------------
extern "C" void kernel_launch(void* const* d_in, const int* in_sizes, int n_in,
                              void* d_out, int out_size, void* d_ws, size_t ws_size,
                              hipStream_t stream) {
    const float* x   = (const float*)d_in[0];
    const int*   src = (const int*)  d_in[1];
    const int*   dst = (const int*)  d_in[2];
    const float* W1  = (const float*)d_in[3];
    const float* al1 = (const float*)d_in[4];
    const float* ar1 = (const float*)d_in[5];
    const float* b1  = (const float*)d_in[6];
    const float* W2  = (const float*)d_in[7];
    const float* al2 = (const float*)d_in[8];
    const float* ar2 = (const float*)d_in[9];
    const float* b2  = (const float*)d_in[10];
    const float* Wf  = (const float*)d_in[11];
    const float* bf  = (const float*)d_in[12];
    float* y = (float*)d_out;

    // Workspace carve-up (256B aligned)
    char* ws = (char*)d_ws;
    size_t off = 0;
    auto carve = [&](size_t bytes) {
        void* p = ws + off;
        off += (bytes + 255) & ~(size_t)255;
        return p;
    };
    float*    hlin  = (float*)    carve((size_t)NNODES * FDIM * 4);  // 51.2 MB
    float*    aggr  = (float*)    carve((size_t)NNODES * FDIM * 4);  // 51.2 MB
    float*    ebuf  = (float*)    carve((size_t)NEDGES * NHEAD * 4); // 25.6 MB
    float*    el    = (float*)    carve((size_t)NNODES * NHEAD * 4);
    float*    er    = (float*)    carve((size_t)NNODES * NHEAD * 4);
    unsigned* maxf  = (unsigned*) carve((size_t)NNODES * NHEAD * 4);
    float*    denom = (float*)    carve((size_t)NNODES * NHEAD * 4);
    _Float16* Wp    = (_Float16*) carve((size_t)FDIM * FDIM * 2);    // 32 KB

    const int gemm_grid = (MTILES + 3) / 4;          // 1563
    const int node_wave = NNODES / 8;                // 12500
    const int nf_grid   = (NNODES * FDIM) / 256;     // 50000
    const int eh_grid   = (NEDGES * NHEAD) / 256;    // 25000
    const int edge_wave = NEDGES / 8;                // 200000

    const float* layer_in[2]  = {x, aggr};
    const float* layer_W[2]   = {W1, W2};
    const float* layer_al[2]  = {al1, al2};
    const float* layer_ar[2]  = {ar1, ar2};
    const float* layer_b[2]   = {b1, b2};

    for (int L = 0; L < 2; ++L) {
        pack_w_kernel<<<64, 256, 0, stream>>>(layer_W[L], Wp);
        gemm_wmma_kernel<<<gemm_grid, 128, 0, stream>>>(layer_in[L], Wp, hlin);
        attn_terms_kernel<<<node_wave, 256, 0, stream>>>(hlin, layer_al[L], layer_ar[L], el, er);
        init_layer_kernel<<<nf_grid, 256, 0, stream>>>(aggr, layer_b[L], maxf, denom);
        edge_logit_max_kernel<<<eh_grid, 256, 0, stream>>>(src, dst, el, er, ebuf, maxf);
        edge_exp_sum_kernel<<<eh_grid, 256, 0, stream>>>(dst, ebuf, maxf, denom);
        edge_aggregate_kernel<<<edge_wave, 256, 0, stream>>>(src, dst, hlin, ebuf, denom, aggr);
    }
    classifier_kernel<<<node_wave, 256, 0, stream>>>(aggr, Wf, bf, y);
}
